// ProposalModule_6193342841692
// MI455X (gfx1250) — matrix operand order repaired and checked by reading, not compiled, not run on hardware
//
#include <hip/hip_runtime.h>

typedef __attribute__((ext_vector_type(16))) _Float16 v16h;
typedef __attribute__((ext_vector_type(8)))  _Float16 v8h;
typedef __attribute__((ext_vector_type(8)))  float    v8f;

#define BATCH 16
#define N1    1024
#define NPQ   256
#define COUT  97
#define N2    2304
#define CF2   351   // comb_feat channels
#define PI_F  3.14159265358979323846f

// ---------------------------------------------------------------------------
// WMMA fragment loaders (CDNA5 16-bit layouts, wave32)
// A (16x32, M x K): lane holds row M=lane&15; halves j -> K = klo+j, klo+16+j
// B (32x16, K x N): lane holds col N=lane&15; halves j -> K = klo+j, klo+16+j
// ---------------------------------------------------------------------------
__device__ __forceinline__ v16h load_fragA(const _Float16* p) {
  v8h lo = *(const v8h*)(p);
  v8h hi = *(const v8h*)(p + 16);
  v16h r;
#pragma unroll
  for (int j = 0; j < 8; ++j) { r[j] = lo[j]; r[j + 8] = hi[j]; }
  return r;
}
__device__ __forceinline__ v16h load_fragB(const _Float16* p) {
  const int l = threadIdx.x & 31;
  const _Float16* b = p - l * 16 + (l & 15);
  const int kh = (l >> 4) << 3;
  v16h r;
#pragma unroll
  for (int j = 0; j < 8; ++j) { r[j] = b[(kh + j) * 16]; r[j + 8] = b[(16 + kh + j) * 16]; }
  return r;
}

// Issue async copy of this lane's 32-byte X row chunk into LDS staging.
// INST_OFFSET is added to BOTH the LDS and global address (ISA 08 §4.4),
// so one operand pair covers both halves. Tracked by ASYNCcnt (+2).
__device__ __forceinline__ void async_stage(unsigned lds_off, const _Float16* gp) {
  asm volatile(
      "global_load_async_to_lds_b128 %0, %1, off\n\t"
      "global_load_async_to_lds_b128 %0, %1, off offset:16"
      :: "v"(lds_off), "v"(gp) : "memory");
}

// One K-step: 1 B-fragment from LDS staging, MT WMMAs against LDS weights.
// A-fragments loaded in batches of 4 so ds_load_b128s pipeline (partial DScnt
// waits) instead of a full s_wait_dscnt 0x0 before every WMMA.
template <int MT>
__device__ __forceinline__ void kstep(v8f (&acc)[MT], const _Float16* abase, int Kpad,
                                      const _Float16* sb) {
  v16h bfrag = load_fragB(sb);
#pragma unroll
  for (int h = 0; h < MT; h += 4) {
    v16h af[4];
#pragma unroll
    for (int q = 0; q < 4 && h + q < MT; ++q)
      af[q] = load_fragA(abase + (size_t)(h + q) * 16 * Kpad);
#pragma unroll
    for (int q = 0; q < 4 && h + q < MT; ++q)
      acc[h + q] = __builtin_amdgcn_wmma_f32_16x16x32_f16(
          false, af[q], false, bfrag, (short)0, acc[h + q], false, false);
  }
}

// ---------------------------------------------------------------------------
// GEMM: Z[p][m] = sum_k W[m][k] * X[k][p] + bias[m]
//   W: f32 [M][K] row-major (global), staged to LDS as f16 [MT*16][Kpad]
//   X: f16 channel-major [Kpad][P]  (rows k >= K are zero)
//   Z: f32 pixel-major  [P][M]
// Block = 256 threads (8 waves); each wave computes one 16-col tile, all M.
// B tiles are double-buffered through LDS with GLOBAL_LOAD_ASYNC_TO_LDS,
// overlapping global latency with the MT-WMMA burst of the current K-step.
// Last K-step peeled so the steady-state loop body is branch-free.
// ---------------------------------------------------------------------------
template <int MT>
__global__ void gemm_wmma(const float* __restrict__ Wt, const float* __restrict__ bias,
                          const _Float16* __restrict__ X, float* __restrict__ Z,
                          int M, int K, int Kpad, int P) {
  extern __shared__ _Float16 wlds[];
  __shared__ _Float16 stage[8 * 2 * 512];  // 8 waves x 2 buffers x (32 lanes x 16 halves)

  const int tid  = threadIdx.x;
  const int rows = MT * 16;
  const int tot  = rows * Kpad;
  for (int i = tid; i < tot; i += 256) {
    int o = i / Kpad, c = i - o * Kpad;
    wlds[i] = (_Float16)((o < M && c < K) ? Wt[o * K + c] : 0.0f);
  }
  __syncthreads();

  const int lane = tid & 31;
  const int wave = tid >> 5;
  const int nlo  = lane & 15;
  const int klo  = (lane >> 4) << 3;   // 0 or 8
  const int NT   = P >> 4;

  _Float16* st = &stage[wave * 1024 + lane * 16];          // buf0; buf1 = +512 halves
  const unsigned so0 = (unsigned)(unsigned long long)st;   // LDS byte offset (addr[31:0])
  const unsigned so1 = so0 + 1024;
  const _Float16* wrow = wlds + (size_t)nlo * Kpad + klo;

  for (int nt = blockIdx.x * 8 + wave; nt < NT; nt += gridDim.x * 8) {
    const int colbase = nt << 4;
    const _Float16* gp = X + (size_t)lane * P + colbase;

    v8f acc[MT];
#pragma unroll
    for (int mt = 0; mt < MT; ++mt)
#pragma unroll
      for (int e = 0; e < 8; ++e) acc[mt][e] = 0.0f;

    async_stage(so0, gp);  // prime buffer 0 with K-step 0

    int kb = 0;
    for (; kb + 32 < Kpad; kb += 32) {            // steady state: branch-free body
      const int buf = (kb >> 5) & 1;
      const _Float16* gpn = gp + (size_t)32 * P;
      async_stage(buf ? so0 : so1, gpn);                 // stage next K-step
      asm volatile("s_wait_asynccnt 0x2" ::: "memory");  // current buffer done
      kstep<MT>(acc, wrow + kb, Kpad, st + buf * 512);
      gp = gpn;
    }
    // peeled last K-step
    asm volatile("s_wait_asynccnt 0x0" ::: "memory");
    kstep<MT>(acc, wrow + kb, Kpad, st + ((kb >> 5) & 1) * 512);

    float* zrow = Z + (size_t)(colbase + nlo) * M;
    const int mhi = klo;  // D layout: lanes 16-31 hold M = v + 8
#pragma unroll
    for (int mt = 0; mt < MT; ++mt)
#pragma unroll
      for (int v = 0; v < 8; ++v) {
        int m = mt * 16 + mhi + v;
        if (m < M) zrow[m] = acc[mt][v] + bias[m];
      }
  }
}

// ---------------------------------------------------------------------------
// Batch-norm helpers: stats over all P pixels, 128 channels
// ---------------------------------------------------------------------------
__global__ void fill_zero(float* p, int n) {
  int i = blockIdx.x * blockDim.x + threadIdx.x;
  if (i < n) p[i] = 0.0f;
}

__global__ void bn_reduce(const float* __restrict__ Z, float* __restrict__ stats,
                          int P, int chunk) {
  int m = threadIdx.x;  // 128 threads
  int p0 = blockIdx.x * chunk;
  int p1 = p0 + chunk; if (p1 > P) p1 = P;
  float s = 0.0f, q = 0.0f;
  for (int p = p0; p < p1; ++p) {
    float v = Z[(size_t)p * 128 + m];
    s += v; q += v * v;
  }
  atomicAdd(&stats[m], s);
  atomicAdd(&stats[128 + m], q);
}

__global__ void bn_finalize(const float* __restrict__ stats, float* __restrict__ musig, int P) {
  int m = threadIdx.x;  // 128
  float invP = 1.0f / (float)P;
  float mu  = stats[m] * invP;
  float var = stats[128 + m] * invP - mu * mu;
  musig[m]       = mu;
  musig[128 + m] = rsqrtf(var + 1e-5f);
}

// y = relu((z-mu)*rsig*g + be) ; output channel-major f16 [128][P]
__global__ void bn_apply(const float* __restrict__ Z, const float* __restrict__ musig,
                         const float* __restrict__ g, const float* __restrict__ be,
                         _Float16* __restrict__ Xo, int P) {
  int i = blockIdx.x * blockDim.x + threadIdx.x;
  if (i >= P * 128) return;
  int m = i & 127, p = i >> 7;
  float v = (Z[i] - musig[m]) * musig[128 + m] * g[m] + be[m];
  Xo[(size_t)m * P + p] = (_Float16)fmaxf(v, 0.0f);
}

// same + max over ns samples per group; output f16 [128][G]
__global__ void bn_apply_maxpool(const float* __restrict__ Z, const float* __restrict__ musig,
                                 const float* __restrict__ g, const float* __restrict__ be,
                                 _Float16* __restrict__ Xo, int G, int ns) {
  int i = blockIdx.x * blockDim.x + threadIdx.x;
  if (i >= G * 128) return;
  int m = i & 127, grp = i >> 7;
  float mu = musig[m], rs = musig[128 + m], gg = g[m], bb = be[m];
  const float* zp = Z + (size_t)grp * ns * 128 + m;
  float best = 0.0f;  // relu(x) >= 0
  for (int k = 0; k < ns; ++k) {
    float v = (zp[(size_t)k * 128] - mu) * rs * gg + bb;
    best = fmaxf(best, fmaxf(v, 0.0f));
  }
  Xo[(size_t)m * G + grp] = (_Float16)best;
}

// ---------------------------------------------------------------------------
// Furthest point sampling: one block per batch, one thread per point
// ---------------------------------------------------------------------------
__global__ void fps_kernel(const float* __restrict__ xyz, int* __restrict__ inds) {
  const int b = blockIdx.x;
  const int tid = threadIdx.x;  // 1024
  __shared__ float s_val[1024];
  __shared__ int   s_idx[1024];
  const float px = xyz[((size_t)b * N1 + tid) * 3 + 0];
  const float py = xyz[((size_t)b * N1 + tid) * 3 + 1];
  const float pz = xyz[((size_t)b * N1 + tid) * 3 + 2];
  float mind = 1e10f;
  int last = 0;
  if (tid == 0) inds[b * NPQ] = 0;
  for (int it = 1; it < NPQ; ++it) {
    float lx = xyz[((size_t)b * N1 + last) * 3 + 0];
    float ly = xyz[((size_t)b * N1 + last) * 3 + 1];
    float lz = xyz[((size_t)b * N1 + last) * 3 + 2];
    float dx = px - lx, dy = py - ly, dz = pz - lz;
    mind = fminf(mind, dx * dx + dy * dy + dz * dz);
    s_val[tid] = mind; s_idx[tid] = tid;
    __syncthreads();
    for (int s = 512; s > 0; s >>= 1) {
      if (tid < s) {
        float v1 = s_val[tid], v2 = s_val[tid + s];
        int i1 = s_idx[tid], i2 = s_idx[tid + s];
        if (v2 > v1 || (v2 == v1 && i2 < i1)) { s_val[tid] = v2; s_idx[tid] = i2; }
      }
      __syncthreads();
    }
    last = s_idx[0];
    __syncthreads();
    if (tid == 0) inds[b * NPQ + it] = last;
  }
}

__global__ void gather_xyz(const float* __restrict__ xyz, const int* __restrict__ inds,
                           float* __restrict__ nx) {
  int i = blockIdx.x * blockDim.x + threadIdx.x;  // B*NPQ*3
  if (i >= BATCH * NPQ * 3) return;
  int d = i % 3, s = (i / 3) % NPQ, b = i / (3 * NPQ);
  nx[i] = xyz[((size_t)b * N1 + inds[b * NPQ + s]) * 3 + d];
}

// features [B][C][N] f32 -> featT [B][N][C] f16
__global__ void feat_transpose(const float* __restrict__ f, _Float16* __restrict__ ft,
                               int B, int C, int N) {
  long long i = (long long)blockIdx.x * blockDim.x + threadIdx.x;
  if (i >= (long long)B * C * N) return;
  int n = (int)(i % N);
  long long t = i / N;
  int c = (int)(t % C), b = (int)(t / C);
  ft[((size_t)b * N + n) * C + c] = (_Float16)f[i];
}

// ---------------------------------------------------------------------------
// Ball query: first ns indices (ascending) with d2 < r2; pad with first hit
// ---------------------------------------------------------------------------
__global__ void ball_select(const float* __restrict__ pts, const float* __restrict__ centers,
                            int* __restrict__ idxout, int B, int N, int S, int ns, float r2) {
  int gid = blockIdx.x * blockDim.x + threadIdx.x;
  if (gid >= B * S) return;
  int b = gid / S;
  float cx = centers[gid * 3 + 0], cy = centers[gid * 3 + 1], cz = centers[gid * 3 + 2];
  const float* P = pts + (size_t)b * N * 3;
  int* out = idxout + (size_t)gid * ns;
  int cnt = 0;
  for (int n = 0; n < N && cnt < ns; ++n) {
    float dx = P[n * 3 + 0] - cx, dy = P[n * 3 + 1] - cy, dz = P[n * 3 + 2] - cz;
    if (dx * dx + dy * dy + dz * dz < r2) out[cnt++] = n;
  }
  int pad = (cnt > 0) ? out[0] : 0;
  for (int k = cnt; k < ns; ++k) out[k] = pad;
}

// Write grouped tensor channel-major: G[c][(g*ns)+k], c = [relxyz(3), feat(Cf), zeros]
__global__ void group_copy(const float* __restrict__ pts, const float* __restrict__ centers,
                           const _Float16* __restrict__ featRows, const int* __restrict__ idxbuf,
                           _Float16* __restrict__ Gout,
                           int B, int N, int S, int ns, int Cf, int Kpad, float invr) {
  int wid  = (blockIdx.x * blockDim.x + threadIdx.x) >> 5;  // one wave per group
  int lane = threadIdx.x & 31;
  if (wid >= B * S) return;
  int b = wid / S;
  const int Ptot = B * S * ns;
  float cx = centers[wid * 3 + 0], cy = centers[wid * 3 + 1], cz = centers[wid * 3 + 2];
  for (int k = 0; k < ns; ++k) {
    int idx = idxbuf[(size_t)wid * ns + k];
    const float* pp = pts + ((size_t)b * N + idx) * 3;
    float rel[3] = { (pp[0] - cx) * invr, (pp[1] - cy) * invr, (pp[2] - cz) * invr };
    const _Float16* frow = featRows + ((size_t)b * N + idx) * Cf;
    int p2 = wid * ns + k;
    for (int c = lane; c < Kpad; c += 32) {
      _Float16 v;
      if (c < 3)            v = (_Float16)rel[c];
      else if (c < 3 + Cf)  v = frow[c - 3];
      else                  v = (_Float16)0.0f;
      Gout[(size_t)c * Ptot + p2] = v;
    }
  }
}

// ---------------------------------------------------------------------------
// Stage-2 combined tensors
// ---------------------------------------------------------------------------
__global__ void build_comb_xyz(const float* __restrict__ nx, const float* __restrict__ xc,
                               const float* __restrict__ xp, float* __restrict__ out) {
  int i = blockIdx.x * blockDim.x + threadIdx.x;  // B*N2*3
  if (i >= BATCH * N2 * 3) return;
  int d = i % 3, n = (i / 3) % N2, b = i / (3 * N2);
  float v;
  if (n < NPQ)           v = nx[((size_t)b * NPQ + n) * 3 + d];
  else if (n < NPQ + N1) v = xc[((size_t)b * N1 + (n - NPQ)) * 3 + d];
  else                   v = xp[((size_t)b * N1 + (n - NPQ - N1)) * 3 + d];
  out[i] = v;
}

// comb_feat [B][N2][CF2] f16: rows 0..255: [net[2:97], feats@inds]; rest: [0(95), corner/plane]
__global__ void build_comb_feat(const float* __restrict__ net, const _Float16* __restrict__ featT,
                                const int* __restrict__ inds, const float* __restrict__ fco,
                                const float* __restrict__ fpl, _Float16* __restrict__ out) {
  long long i = (long long)blockIdx.x * blockDim.x + threadIdx.x;
  if (i >= (long long)BATCH * N2 * CF2) return;
  int c = (int)(i % CF2);
  long long t = i / CF2;
  int n = (int)(t % N2), b = (int)(t / N2);
  float v;
  if (n < NPQ) {
    if (c < COUT - 2) v = net[((size_t)(b * NPQ + n)) * COUT + (c + 2)];
    else {
      int idx = inds[b * NPQ + n];
      v = (float)featT[((size_t)b * N1 + idx) * 256 + (c - (COUT - 2))];
    }
  } else if (n < NPQ + N1) {
    v = (c < COUT - 2) ? 0.0f : fco[((size_t)b * 256 + (c - (COUT - 2))) * N1 + (n - NPQ)];
  } else {
    v = (c < COUT - 2) ? 0.0f : fpl[((size_t)b * 256 + (c - (COUT - 2))) * N1 + (n - NPQ - N1)];
  }
  out[i] = (_Float16)v;
}

// ---------------------------------------------------------------------------
// Decode: channel layout [obj(2) center(3) hs(1) hr(1) ss(18) sr(54) sem(18)]
// ---------------------------------------------------------------------------
__device__ __forceinline__ float decode_val(float tv, int c, int gs,
                                            const float* nx, const float* msize) {
  if (c < 2)  return tv;
  if (c < 5)  return nx[gs * 3 + (c - 2)] + tv;
  if (c < 6)  return tv;            // heading score
  if (c < 7)  return tv * PI_F;     // heading residual * (pi/NH), NH=1
  if (c < 25) return tv;            // size scores
  if (c < 79) { int j = (c - 25) / 3, d = (c - 25) % 3; return tv * msize[j * 3 + d]; }
  return tv;                        // semantic
}

__global__ void decode1(const float* __restrict__ net, const float* __restrict__ nx,
                        const float* __restrict__ msize, float* __restrict__ out) {
  int i = blockIdx.x * blockDim.x + threadIdx.x;  // 4096*97
  if (i >= BATCH * NPQ * COUT) return;
  int c = i % COUT, gs = i / COUT;
  out[i] = decode_val(net[i], c, gs, nx, msize);
}

__global__ void decode2(const float* __restrict__ net, const float* __restrict__ nc,
                        const float* __restrict__ nx, const float* __restrict__ msize,
                        float* __restrict__ out) {
  int i = blockIdx.x * blockDim.x + threadIdx.x;
  if (i >= BATCH * NPQ * COUT) return;
  int c = i % COUT, gs = i / COUT;
  float tv = (c < 2) ? net[(size_t)gs * COUT + c] : nc[(size_t)gs * (COUT - 2) + (c - 2)];
  out[i] = decode_val(tv, c, gs, nx, msize);
}

// ---------------------------------------------------------------------------
// Host orchestration
// ---------------------------------------------------------------------------
extern "C" void kernel_launch(void* const* d_in, const int* in_sizes, int n_in,
                              void* d_out, int out_size, void* d_ws, size_t ws_size,
                              hipStream_t stream) {
  (void)in_sizes; (void)n_in; (void)out_size; (void)ws_size;
  const float* xyz    = (const float*)d_in[0];
  const float* feats  = (const float*)d_in[1];
  const float* xyz_c  = (const float*)d_in[2];
  const float* feat_c = (const float*)d_in[3];
  const float* xyz_p  = (const float*)d_in[4];
  const float* feat_p = (const float*)d_in[5];
  // params flattened in insertion order: sa1(3x4), sa2(3x4), head(10), corner(10), mean_size
  const float* sa1W[3], *sa1b[3], *sa1g[3], *sa1be[3];
  const float* sa2W[3], *sa2b[3], *sa2g[3], *sa2be[3];
  for (int i = 0; i < 3; ++i) {
    sa1W[i]  = (const float*)d_in[6 + 4 * i + 0];
    sa1b[i]  = (const float*)d_in[6 + 4 * i + 1];
    sa1g[i]  = (const float*)d_in[6 + 4 * i + 2];
    sa1be[i] = (const float*)d_in[6 + 4 * i + 3];
    sa2W[i]  = (const float*)d_in[18 + 4 * i + 0];
    sa2b[i]  = (const float*)d_in[18 + 4 * i + 1];
    sa2g[i]  = (const float*)d_in[18 + 4 * i + 2];
    sa2be[i] = (const float*)d_in[18 + 4 * i + 3];
  }
  const float* hW1 = (const float*)d_in[30], *hb1 = (const float*)d_in[31];
  const float* hg1 = (const float*)d_in[32], *hbe1 = (const float*)d_in[33];
  const float* hW2 = (const float*)d_in[34], *hb2 = (const float*)d_in[35];
  const float* hg2 = (const float*)d_in[36], *hbe2 = (const float*)d_in[37];
  const float* hW3 = (const float*)d_in[38], *hb3 = (const float*)d_in[39];
  const float* cW1 = (const float*)d_in[40], *cb1 = (const float*)d_in[41];
  const float* cg1 = (const float*)d_in[42], *cbe1 = (const float*)d_in[43];
  const float* cW2 = (const float*)d_in[44], *cb2 = (const float*)d_in[45];
  const float* cg2 = (const float*)d_in[46], *cbe2 = (const float*)d_in[47];
  const float* cW3 = (const float*)d_in[48], *cb3 = (const float*)d_in[49];
  const float* msize = (const float*)d_in[50];

  // ---- workspace carve-up ----
  char* base = (char*)d_ws;
  size_t off = 0;
  auto alloc = [&](size_t bytes) -> char* {
    off = (off + 255) & ~(size_t)255;
    char* p = base + off; off += bytes; return p;
  };
  const int G = BATCH * NPQ;      // 4096 groups
  const int P1 = G * 16;          // 65536 stage-1 pixels
  const int P2 = G * 32;          // 131072 stage-2 pixels
  int*      inds  = (int*)      alloc((size_t)G * 4);
  float*    nx    = (float*)    alloc((size_t)G * 3 * 4);
  _Float16* featT = (_Float16*) alloc((size_t)BATCH * N1 * 256 * 2);
  int*      idx1  = (int*)      alloc((size_t)G * 16 * 4);
  int*      idx2  = (int*)      alloc((size_t)G * 32 * 4);
  float*    stats = (float*)    alloc(256 * 4);
  float*    musig = (float*)    alloc(256 * 4);
  _Float16* feat1 = (_Float16*) alloc((size_t)128 * G * 2);
  _Float16* feat2 = (_Float16*) alloc((size_t)128 * G * 2);
  float*    hz    = (float*)    alloc((size_t)G * 128 * 4);
  _Float16* hxA   = (_Float16*) alloc((size_t)128 * G * 2);
  _Float16* hxB   = (_Float16*) alloc((size_t)128 * G * 2);
  float*    net   = (float*)    alloc((size_t)G * COUT * 4);
  float*    nc    = (float*)    alloc((size_t)G * (COUT - 2) * 4);
  float*    cxyz  = (float*)    alloc((size_t)BATCH * N2 * 3 * 4);
  _Float16* cfeat = (_Float16*) alloc((size_t)BATCH * N2 * CF2 * 2);
  char*     arena = alloc((size_t)234881024);
  // stage-1 views (dead before stage-2 writes begin)
  _Float16* g1  = (_Float16*)(arena);
  float*    z1  = (float*)   (arena + 37748736);
  _Float16* x1A = (_Float16*)(arena + 71303168);
  _Float16* x1B = (_Float16*)(arena + 88080384);
  // stage-2 views
  _Float16* g2  = (_Float16*)(arena);
  float*    z2  = (float*)   (arena + 100663296);
  _Float16* x2A = (_Float16*)(arena + 167772160);
  _Float16* x2B = (_Float16*)(arena + 201326592);

  auto gemm128 = [&](const float* W, const float* b, const _Float16* X, float* Z,
                     int K, int Kpad, int P) {
    int blocks = (P / 16 + 7) / 8;
    size_t smem = (size_t)128 * Kpad * 2;
    gemm_wmma<8><<<blocks, 256, smem, stream>>>(W, b, X, Z, 128, K, Kpad, P);
  };
  auto run_bn = [&](const float* Z, const float* g, const float* be, _Float16* Xo, int P) {
    fill_zero<<<1, 256, 0, stream>>>(stats, 256);
    int chunk = 512;
    bn_reduce<<<(P + chunk - 1) / chunk, 128, 0, stream>>>(Z, stats, P, chunk);
    bn_finalize<<<1, 128, 0, stream>>>(stats, musig, P);
    bn_apply<<<(P * 128 + 255) / 256, 256, 0, stream>>>(Z, musig, g, be, Xo, P);
  };
  auto run_bn_max = [&](const float* Z, const float* g, const float* be, _Float16* Xo, int ns) {
    int P = G * ns;
    fill_zero<<<1, 256, 0, stream>>>(stats, 256);
    int chunk = 512;
    bn_reduce<<<(P + chunk - 1) / chunk, 128, 0, stream>>>(Z, stats, P, chunk);
    bn_finalize<<<1, 128, 0, stream>>>(stats, musig, P);
    bn_apply_maxpool<<<(G * 128 + 255) / 256, 256, 0, stream>>>(Z, musig, g, be, Xo, G, ns);
  };

  // ---- stage 1 ----
  fps_kernel<<<BATCH, 1024, 0, stream>>>(xyz, inds);
  gather_xyz<<<(G * 3 + 255) / 256, 256, 0, stream>>>(xyz, inds, nx);
  {
    long long tot = (long long)BATCH * 256 * N1;
    feat_transpose<<<(unsigned)((tot + 255) / 256), 256, 0, stream>>>(feats, featT, BATCH, 256, N1);
  }
  ball_select<<<(G + 255) / 256, 256, 0, stream>>>(xyz, nx, idx1, BATCH, N1, NPQ, 16, 0.09f);
  group_copy<<<G / 8, 256, 0, stream>>>(xyz, nx, featT, idx1, g1,
                                        BATCH, N1, NPQ, 16, 256, 288, 1.0f / 0.3f);
  gemm128(sa1W[0], sa1b[0], g1, z1, 259, 288, P1);
  run_bn(z1, sa1g[0], sa1be[0], x1A, P1);
  gemm128(sa1W[1], sa1b[1], x1A, z1, 128, 128, P1);
  run_bn(z1, sa1g[1], sa1be[1], x1B, P1);
  gemm128(sa1W[2], sa1b[2], x1B, z1, 128, 128, P1);
  run_bn_max(z1, sa1g[2], sa1be[2], feat1, 16);
  // head
  gemm128(hW1, hb1, feat1, hz, 128, 128, G);
  run_bn(hz, hg1, hbe1, hxA, G);
  gemm128(hW2, hb2, hxA, hz, 128, 128, G);
  run_bn(hz, hg2, hbe2, hxB, G);
  {
    int blocks = (G / 16 + 7) / 8;
    size_t smem = (size_t)7 * 16 * 128 * 2;
    gemm_wmma<7><<<blocks, 256, smem, stream>>>(hW3, hb3, hxB, net, COUT, 128, 128, G);
  }
  decode1<<<(G * COUT + 255) / 256, 256, 0, stream>>>(net, nx, msize, (float*)d_out);

  // ---- stage 2 ----
  build_comb_xyz<<<(BATCH * N2 * 3 + 255) / 256, 256, 0, stream>>>(nx, xyz_c, xyz_p, cxyz);
  {
    long long tot = (long long)BATCH * N2 * CF2;
    build_comb_feat<<<(unsigned)((tot + 255) / 256), 256, 0, stream>>>(net, featT, inds,
                                                                      feat_c, feat_p, cfeat);
  }
  ball_select<<<(G + 255) / 256, 256, 0, stream>>>(cxyz, nx, idx2, BATCH, N2, NPQ, 32, 0.36f);
  group_copy<<<G / 8, 256, 0, stream>>>(cxyz, nx, cfeat, idx2, g2,
                                        BATCH, N2, NPQ, 32, CF2, 384, 1.0f / 0.6f);
  gemm128(sa2W[0], sa2b[0], g2, z2, 354, 384, P2);
  run_bn(z2, sa2g[0], sa2be[0], x2A, P2);
  gemm128(sa2W[1], sa2b[1], x2A, z2, 128, 128, P2);
  run_bn(z2, sa2g[1], sa2be[1], x2B, P2);
  gemm128(sa2W[2], sa2b[2], x2B, z2, 128, 128, P2);
  run_bn_max(z2, sa2g[2], sa2be[2], feat2, 32);
  // corner head
  gemm128(cW1, cb1, feat2, hz, 128, 128, G);
  run_bn(hz, cg1, cbe1, hxA, G);
  gemm128(cW2, cb2, hxA, hz, 128, 128, G);
  run_bn(hz, cg2, cbe2, hxB, G);
  {
    int blocks = (G / 16 + 7) / 8;
    size_t smem = (size_t)6 * 16 * 128 * 2;
    gemm_wmma<6><<<blocks, 256, smem, stream>>>(cW3, cb3, hxB, nc, COUT - 2, 128, 128, G);
  }
  decode2<<<(G * COUT + 255) / 256, 256, 0, stream>>>(net, nc, nx, msize,
                                                      (float*)d_out + (size_t)G * COUT);
}
